// FlashFFN_79345225826315
// MI455X (gfx1250) — matrix operand order repaired and checked by reading, not compiled, not run on hardware
//
#include <hip/hip_runtime.h>
#include <stdint.h>

typedef __attribute__((ext_vector_type(16))) __bf16 v16bf;
typedef __attribute__((ext_vector_type(8)))  float  v8f;

#define MDIM 131072
#define KDIM 1024
#define NDIM 512
#define BM   32
#define X1_STRIDE 1032   // 1024 + 8 pad (bf16), keeps 16B align, rotates banks
#define X3_STRIDE 520    // 512 + 8 pad

__device__ __forceinline__ unsigned short f2bf(float f) {
    uint32_t u = __builtin_bit_cast(uint32_t, f);
    u += 0x7FFFu + ((u >> 16) & 1u);          // round-to-nearest-even
    return (unsigned short)(u >> 16);
}

union Frag { v16bf bf; uint4 q[2]; };

// A-fragment (16x32 bf16, ISA 7.12.2): lane<16 -> K = kbase+{0..7, 16..23}
__device__ __forceinline__ v16bf load_frag_a_lds(const unsigned short* lds,
                                                 int row, int stride, int kbase) {
    Frag f;
    const uint4* p = reinterpret_cast<const uint4*>(lds + row * stride + kbase);
    f.q[0] = p[0];   // elements kbase .. kbase+7
    f.q[1] = p[2];   // elements kbase+16 .. kbase+23
    return f.bf;
}

// B-fragment: weights pre-swizzled so each lane's 16 bf16 are contiguous (32B).
__device__ __forceinline__ v16bf load_frag_b(const unsigned short* pw,
                                             int tile, int lane) {
    Frag f;
    const uint4* p = reinterpret_cast<const uint4*>(pw + (size_t)tile * 512 + lane * 16);
    f.q[0] = p[0];
    f.q[1] = p[1];
    return f.bf;
}

#define WMMA_BF16(A, B, C) \
    __builtin_amdgcn_wmma_f32_16x16x32_bf16(false, (A), false, (B), (short)0, (C), false, false)

// ---------------------------------------------------------------------------
// Kernel 1: fp32 -> bf16 weight conversion, pre-swizzled into WMMA B layout.
// Packed layout: p[tile][lane][elem], tile = kb*(Ncols/16)+nt,
// krow = kb*32 + (lane>=16 ? 8 : 0) + (elem<8 ? elem : elem+8), col = nt*16 + (lane&15).
// ---------------------------------------------------------------------------
__global__ void __launch_bounds__(256) pack_weights_kernel(
    const float* __restrict__ w1, const float* __restrict__ w2,
    const float* __restrict__ w3,
    unsigned short* __restrict__ p1, unsigned short* __restrict__ p2,
    unsigned short* __restrict__ p3)
{
    const int e     = blockIdx.x * 256 + threadIdx.x;   // 0 .. 524287
    const int which = blockIdx.y;                       // 0:w1 1:w2 2:w3
    const int tile  = e >> 9;
    const int r     = e & 511;
    const int lane  = r >> 4;
    const int elem  = r & 15;
    const int khalf = (lane >= 16) ? 8 : 0;
    const int nlow  = lane & 15;

    if (which < 2) {
        const float* w        = which ? w2 : w1;
        unsigned short* p     = which ? p2 : p1;
        const int kb   = tile >> 5;          // K=1024 -> 32 K-blocks
        const int nt   = tile & 31;          // N=512  -> 32 N-tiles
        const int krow = kb * 32 + khalf + (elem < 8 ? elem : elem + 8);
        const int col  = nt * 16 + nlow;
        p[e] = f2bf(w[(size_t)krow * NDIM + col]);
    } else {
        const int kb   = tile >> 6;          // K=512 -> 16 K-blocks
        const int nt   = tile & 63;          // N=1024 -> 64 N-tiles
        const int krow = kb * 32 + khalf + (elem < 8 ? elem : elem + 8);
        const int col  = nt * 16 + nlow;
        p3[e] = f2bf(w3[(size_t)krow * KDIM + col]);
    }
}

// ---------------------------------------------------------------------------
// Kernel 2: fused RMSNorm -> (gate,up) GEMM -> SiLU -> down GEMM.
// 1 block = 32 rows, 256 threads = 8 wave32.
// B fragments are double-buffered: iteration kb consumes fragments issued in
// iteration kb-1, so the pre-WMMA wait is loadcnt<=8 instead of 0.
// ---------------------------------------------------------------------------
__global__ void __launch_bounds__(256) ffn_kernel(
    const float* __restrict__ x, const float* __restrict__ w0,
    const unsigned short* __restrict__ p1, const unsigned short* __restrict__ p2,
    const unsigned short* __restrict__ p3, float* __restrict__ out)
{
    __shared__ __align__(16) unsigned short lds_x1[BM * X1_STRIDE]; // 66048 B
    __shared__ __align__(16) unsigned short lds_x3[BM * X3_STRIDE]; // 33280 B

    const int tid  = threadIdx.x;
    const int lane = tid & 31;
    const int wave = tid >> 5;
    const int row0 = blockIdx.x * BM;

    // ---------- Phase 1: RMSNorm + bf16 convert into LDS ----------
    {
        const int r   = tid >> 3;      // 0..31, row within block
        const int sub = tid & 7;       // 8 lanes per row
        const float4* xrow4 = reinterpret_cast<const float4*>(x + (size_t)(row0 + r) * KDIM);
        const float4* w04   = reinterpret_cast<const float4*>(w0);

        float ss = 0.f;
        #pragma unroll
        for (int i = 0; i < 32; ++i) {
            float4 v = xrow4[sub + i * 8];
            ss += v.x * v.x + v.y * v.y + v.z * v.z + v.w * v.w;
        }
        ss += __shfl_xor(ss, 1, 32);
        ss += __shfl_xor(ss, 2, 32);
        ss += __shfl_xor(ss, 4, 32);
        const float l = rsqrtf(ss * (1.0f / (float)KDIM));

        unsigned short* dst = lds_x1 + r * X1_STRIDE;
        #pragma unroll
        for (int i = 0; i < 32; ++i) {
            const int c4 = sub + i * 8;
            float4 v = xrow4[c4];
            float4 g = w04[c4];
            ushort4 o;
            o.x = f2bf(v.x * l * g.x);
            o.y = f2bf(v.y * l * g.y);
            o.z = f2bf(v.z * l * g.z);
            o.w = f2bf(v.w * l * g.w);
            reinterpret_cast<ushort4*>(dst)[c4] = o;
        }
    }
    __syncthreads();

    // ---------- Phase 2: gate = x1@w1, up = x1@w2, SiLU-combine -> lds_x3 ----------
    {
        const int mt     = wave >> 2;            // tile-row 0..1
        const int ntbase = (wave & 3) * 8;       // 8 N-tiles per wave
        const int arow   = mt * 16 + (lane & 15);
        const int koff   = (lane >= 16) ? 8 : 0; // A-fragment half-K select
        const int cbase  = lane & 15;
        const int rext   = (lane >= 16) ? 8 : 0; // C-fragment row extension

        #pragma unroll
        for (int pr = 0; pr < 4; ++pr) {
            const int ntA = ntbase + 2 * pr;
            const int ntB = ntA + 1;
            v8f g0 = {}, u0 = {}, g1 = {}, u1 = {};

            // Prologue: B fragments for kb=0.
            v16bf b0 = load_frag_b(p1, ntA, lane);
            v16bf b1 = load_frag_b(p2, ntA, lane);
            v16bf b2 = load_frag_b(p1, ntB, lane);
            v16bf b3 = load_frag_b(p2, ntB, lane);

            for (int kb = 0; kb < 32; ++kb) {
                const int kn = (kb + 1 < 32) ? kb + 1 : kb;   // last: harmless replay
                const v16bf a = load_frag_a_lds(lds_x1, arow, X1_STRIDE, kb * 32 + koff);
                // Issue next iteration's B fragments before consuming current ones.
                const v16bf n0 = load_frag_b(p1, kn * 32 + ntA, lane);
                const v16bf n1 = load_frag_b(p2, kn * 32 + ntA, lane);
                const v16bf n2 = load_frag_b(p1, kn * 32 + ntB, lane);
                const v16bf n3 = load_frag_b(p2, kn * 32 + ntB, lane);
                if (kb + 2 < 32) {  // WGP-scope prefetch (pull into L0), 2 ahead
                    __builtin_prefetch(p1 + ((size_t)((kb + 2) * 32 + ntA) * 512), 0, 3);
                    __builtin_prefetch(p2 + ((size_t)((kb + 2) * 32 + ntA) * 512), 0, 3);
                }
                g0 = WMMA_BF16(a, b0, g0);
                u0 = WMMA_BF16(a, b1, u0);
                g1 = WMMA_BF16(a, b2, g1);
                u1 = WMMA_BF16(a, b3, u1);
                b0 = n0; b1 = n1; b2 = n2; b3 = n3;
            }

            #pragma unroll
            for (int v = 0; v < 8; ++v) {
                const int mr = mt * 16 + v + rext;
                const float ga = g0[v];
                const float gb = g1[v];
                const float sa = ga / (1.0f + __expf(-ga));
                const float sb = gb / (1.0f + __expf(-gb));
                lds_x3[mr * X3_STRIDE + ntA * 16 + cbase] = f2bf(sa * u0[v]);
                lds_x3[mr * X3_STRIDE + ntB * 16 + cbase] = f2bf(sb * u1[v]);
            }
        }
    }
    __syncthreads();

    // ---------- Phase 3: out = x3 @ w3 ----------
    {
        const int mt     = wave >> 2;
        const int ntbase = (wave & 3) * 16;      // 16 N-tiles (of 64) per wave
        const int arow   = mt * 16 + (lane & 15);
        const int koff   = (lane >= 16) ? 8 : 0;
        const int cbase  = lane & 15;
        const int rext   = (lane >= 16) ? 8 : 0;

        #pragma unroll
        for (int g = 0; g < 4; ++g) {
            v8f acc0 = {}, acc1 = {}, acc2 = {}, acc3 = {};
            const int nt0 = ntbase + g * 4;

            v16bf b0 = load_frag_b(p3, nt0 + 0, lane);
            v16bf b1 = load_frag_b(p3, nt0 + 1, lane);
            v16bf b2 = load_frag_b(p3, nt0 + 2, lane);
            v16bf b3 = load_frag_b(p3, nt0 + 3, lane);

            for (int kb = 0; kb < 16; ++kb) {
                const int kn = (kb + 1 < 16) ? kb + 1 : kb;
                const v16bf a = load_frag_a_lds(lds_x3, arow, X3_STRIDE, kb * 32 + koff);
                const v16bf n0 = load_frag_b(p3, kn * 64 + nt0 + 0, lane);
                const v16bf n1 = load_frag_b(p3, kn * 64 + nt0 + 1, lane);
                const v16bf n2 = load_frag_b(p3, kn * 64 + nt0 + 2, lane);
                const v16bf n3 = load_frag_b(p3, kn * 64 + nt0 + 3, lane);
                if (kb + 2 < 16)
                    __builtin_prefetch(p3 + ((size_t)((kb + 2) * 64 + nt0) * 512), 0, 3);
                acc0 = WMMA_BF16(a, b0, acc0);
                acc1 = WMMA_BF16(a, b1, acc1);
                acc2 = WMMA_BF16(a, b2, acc2);
                acc3 = WMMA_BF16(a, b3, acc3);
                b0 = n0; b1 = n1; b2 = n2; b3 = n3;
            }

            #pragma unroll
            for (int v = 0; v < 8; ++v) {
                const size_t r = (size_t)(row0 + mt * 16 + v + rext) * KDIM;
                out[r + (nt0 + 0) * 16 + cbase] = acc0[v];
                out[r + (nt0 + 1) * 16 + cbase] = acc1[v];
                out[r + (nt0 + 2) * 16 + cbase] = acc2[v];
                out[r + (nt0 + 3) * 16 + cbase] = acc3[v];
            }
        }
    }
}

extern "C" void kernel_launch(void* const* d_in, const int* in_sizes, int n_in,
                              void* d_out, int out_size, void* d_ws, size_t ws_size,
                              hipStream_t stream) {
    const float* x  = (const float*)d_in[0];
    const float* w0 = (const float*)d_in[1];
    const float* w1 = (const float*)d_in[2];
    const float* w2 = (const float*)d_in[3];
    const float* w3 = (const float*)d_in[4];
    float* out = (float*)d_out;

    unsigned short* p1 = (unsigned short*)d_ws;                 // 1 MB each
    unsigned short* p2 = p1 + (size_t)KDIM * NDIM;
    unsigned short* p3 = p2 + (size_t)KDIM * NDIM;

    // Pack/convert weights once (3 x 524288 bf16 elems).
    pack_weights_kernel<<<dim3(2048, 3, 1), 256, 0, stream>>>(w1, w2, w3, p1, p2, p3);

    // Fused FFN: 4096 blocks x 256 threads, 32 rows per block.
    ffn_kernel<<<dim3(MDIM / BM, 1, 1), 256, 0, stream>>>(x, w0, p1, p2, p3, out);
}
